// SAGE_GAT_70772471103693
// MI455X (gfx1250) — matrix operand order repaired and checked by reading, not compiled, not run on hardware
//
#include <hip/hip_runtime.h>
#include <hip/hip_bf16.h>
#include <math.h>

// ---------------------------------------------------------------------------
// SAGEConv + GATConv pipeline for gfx1250 (MI455X, wave32, WMMA).
//  - Dense GEMMs: v_wmma_f32_16x16x32_f16, weights pre-packed to f16 fragment
//    layout once, staged per-block into LDS with global_load_async_to_lds_b128
//    (ASYNCcnt) and read back as ds_load_b128.
//  - Irregular parts: native float atomics + ordered-uint atomicMax.
// ---------------------------------------------------------------------------

typedef __attribute__((ext_vector_type(16))) _Float16 v16h;
typedef __attribute__((ext_vector_type(8)))  _Float16 v8h;
typedef __attribute__((ext_vector_type(8)))  float    v8f;

#define F_IN  64
#define HDIM  64
#define CDIM  16
#define NEG_SLOPE 0.2f

// ---- gfx1250 async copy to LDS (inline asm; ASYNCcnt tracked) -------------
__device__ __forceinline__ void async_copy_b128(unsigned ldsAddr, const void* gptr) {
    unsigned long long ga = (unsigned long long)(size_t)gptr;
    asm volatile("global_load_async_to_lds_b128 %0, %1, off"
                 :: "v"(ldsAddr), "v"(ga) : "memory");
}
__device__ __forceinline__ void wait_async0() {
    asm volatile("s_wait_asynccnt 0x0" ::: "memory");
}
// Truncation of a flat LDS pointer to 32 bits yields the LDS byte address
// (ISA 10.2: LDS aperture; the low 32 address bits are the LDS offset).
__device__ __forceinline__ unsigned lds_addr(const void* p) {
    return (unsigned)(size_t)p;
}

// Monotonic float<->uint mapping so atomicMax(u32) implements float max.
__device__ __forceinline__ unsigned order_f32(float f) {
    unsigned b = __float_as_uint(f);
    return (b & 0x80000000u) ? ~b : (b | 0x80000000u);
}
__device__ __forceinline__ float unorder_f32(unsigned u) {
    unsigned b = (u & 0x80000000u) ? (u & 0x7FFFFFFFu) : ~u;
    return __uint_as_float(b);
}

// ---------------------------------------------------------------------------
// Prep A: repack weights f32 -> f16 in WMMA B-fragment order.
// Fragment storage: idx(kt,ct,lane,e) = (((kt*nct+ct)*32)+lane)*16 + e
// maps to W[k*ncols+col] with k = kt*32 + (lane>=16)*16 + e, col = ct*16 + (lane&15).
// wl/wr: K=64,ncols=64 (4096 halves each). wg: K=64,ncols=16 (1024 halves).
// ---------------------------------------------------------------------------
__global__ void k_prep_w(const float* __restrict__ wl, const float* __restrict__ wr,
                         const float* __restrict__ wg, _Float16* __restrict__ wlf,
                         _Float16* __restrict__ wrf, _Float16* __restrict__ wgf) {
    int id = blockIdx.x * blockDim.x + threadIdx.x;
    if (id < 8192) {
        const float* W = (id < 4096) ? wl : wr;
        _Float16*    O = (id < 4096) ? wlf : wrf;
        int fid  = id & 4095;
        int e    = fid & 15;
        int lane = (fid >> 4) & 31;
        int ct   = (fid >> 9) & 3;
        int kt   = fid >> 11;
        int k    = kt * 32 + (lane >> 4) * 16 + e;
        int col  = ct * 16 + (lane & 15);
        O[fid] = (_Float16)W[k * HDIM + col];
    } else if (id < 8192 + 1024) {
        int fid  = id - 8192;
        int e    = fid & 15;
        int lane = (fid >> 4) & 31;
        int kt   = fid >> 9;
        int k    = kt * 32 + (lane >> 4) * 16 + e;
        int col  = lane & 15;
        wgf[fid] = (_Float16)wg[k * CDIM + col];
    }
}

// ---------------------------------------------------------------------------
// Stage 1: scatter x[src] into agg[dst] + degree count (global_atomic_add_f32).
// ---------------------------------------------------------------------------
__global__ void k_scatter(const float* __restrict__ x,
                          const int* __restrict__ src,
                          const int* __restrict__ dst,
                          float* __restrict__ agg,
                          float* __restrict__ deg, int E) {
    long long gid = (long long)blockIdx.x * blockDim.x + threadIdx.x;
    if (gid >= (long long)E * F_IN) return;
    int e = (int)(gid >> 6);
    int f = (int)(gid & 63);
    __builtin_prefetch(&src[e + 4096], 0, 0);   // global_prefetch_b8 (speculative)
    __builtin_prefetch(&dst[e + 4096], 0, 0);
    int s = src[e], d = dst[e];
    unsafeAtomicAdd(&agg[(size_t)d * F_IN + f], x[(size_t)s * F_IN + f]);
    if (f == 0) unsafeAtomicAdd(&deg[d], 1.0f);
}

// ---------------------------------------------------------------------------
// Prep B: xh = f16(x), meanh = f16(agg/deg).  (feeds WMMA A-fragments directly)
// ---------------------------------------------------------------------------
__global__ void k_prep_feat(const float* __restrict__ x, const float* __restrict__ agg,
                            const float* __restrict__ deg, _Float16* __restrict__ xh,
                            _Float16* __restrict__ meanh, int N) {
    long long id = (long long)blockIdx.x * blockDim.x + threadIdx.x;
    if (id >= (long long)N * F_IN) return;
    int node = (int)(id >> 6);
    float inv = 1.0f / fmaxf(deg[node], 1.0f);
    xh[id]    = (_Float16)x[id];
    meanh[id] = (_Float16)(agg[id] * inv);
}

// Build a 16-half A fragment from two aligned 8-half chunks of a row-major row.
__device__ __forceinline__ v16h a_frag(const _Float16* row, int k0, int kA) {
    v8h lo = *(const v8h*)(row + k0 + kA);
    v8h hi = *(const v8h*)(row + k0 + kA + 8);
    return __builtin_shufflevector(lo, hi, 0, 1, 2, 3, 4, 5, 6, 7,
                                           8, 9, 10, 11, 12, 13, 14, 15);
}

// ---------------------------------------------------------------------------
// Stage 2: h = relu(mean @ Wl + bl + x @ Wr).  4 waves/block, 16 rows/wave.
// Weights async-staged in LDS (fragment layout), read via ds_load_b128.
// ---------------------------------------------------------------------------
__global__ void __launch_bounds__(128)
k_sage_wmma(const _Float16* __restrict__ xh, const _Float16* __restrict__ meanh,
            const _Float16* __restrict__ wlf, const _Float16* __restrict__ wrf,
            const float* __restrict__ bl, _Float16* __restrict__ h, int N) {
    __shared__ _Float16 sWl[4096];   // 8KB
    __shared__ _Float16 sWr[4096];   // 8KB
    const int tid = threadIdx.x;

    // 512 x 16B chunks per matrix, 128 threads -> 4 iterations, uniform.
    for (int i = tid; i < 512; i += 128) {
        async_copy_b128(lds_addr(&sWl[i * 8]), wlf + i * 8);
        async_copy_b128(lds_addr(&sWr[i * 8]), wrf + i * 8);
    }
    wait_async0();
    __syncthreads();

    const int lane    = tid & 31;
    const int wave    = tid >> 5;
    const int rowBase = blockIdx.x * 64 + wave * 16;
    int r = rowBase + (lane & 15);
    if (r >= N) r = N - 1;                  // clamp loads; stores guarded below
    const int kA = (lane >> 4) * 16;
    const int n  = lane & 15;
    const _Float16* xrow = xh    + (size_t)r * HDIM;
    const _Float16* mrow = meanh + (size_t)r * HDIM;

    v8f acc[4] = {{0.f,0.f,0.f,0.f,0.f,0.f,0.f,0.f}};

#pragma unroll
    for (int kt = 0; kt < 2; ++kt) {
        const int k0 = kt * 32;
        v16h am = a_frag(mrow, k0, kA);
        v16h ax = a_frag(xrow, k0, kA);
#pragma unroll
        for (int ct = 0; ct < 4; ++ct) {
            const int fidx = ((kt * 4 + ct) * 32 + lane) * 16;
            v16h bml = *(const v16h*)&sWl[fidx];
            v16h bxr = *(const v16h*)&sWr[fidx];
            acc[ct] = __builtin_amdgcn_wmma_f32_16x16x32_f16(
                false, am, false, bml, (short)0, acc[ct], false, false);
            acc[ct] = __builtin_amdgcn_wmma_f32_16x16x32_f16(
                false, ax, false, bxr, (short)0, acc[ct], false, false);
        }
    }

    const int mofs = (lane >> 4) * 8;
#pragma unroll
    for (int ct = 0; ct < 4; ++ct) {
        const int col  = ct * 16 + n;
        const float bb = bl[col];
#pragma unroll
        for (int v = 0; v < 8; ++v) {
            const int row = rowBase + mofs + v;
            if (row < N) {
                float val = fmaxf(acc[ct][v] + bb, 0.0f);   // ReLU
                h[(size_t)row * HDIM + col] = (_Float16)val;
            }
        }
    }
}

// ---------------------------------------------------------------------------
// Stage 3: g = h @ w_gat  ([N,64] f16 x [64,16] -> [N,16] f32).
// ---------------------------------------------------------------------------
__global__ void __launch_bounds__(128)
k_gat_wmma(const _Float16* __restrict__ h, const _Float16* __restrict__ wgf,
           float* __restrict__ g, int N) {
    __shared__ _Float16 sWg[1024];   // 2KB = 128 x 16B chunks
    const int tid = threadIdx.x;
    async_copy_b128(lds_addr(&sWg[tid * 8]), wgf + tid * 8);
    wait_async0();
    __syncthreads();

    const int lane    = tid & 31;
    const int wave    = tid >> 5;
    const int rowBase = blockIdx.x * 64 + wave * 16;
    int r = rowBase + (lane & 15);
    if (r >= N) r = N - 1;
    const int kA = (lane >> 4) * 16;
    const int n  = lane & 15;
    const _Float16* hrow = h + (size_t)r * HDIM;

    v8f acc = {0.f,0.f,0.f,0.f,0.f,0.f,0.f,0.f};
#pragma unroll
    for (int kt = 0; kt < 2; ++kt) {
        v16h a = a_frag(hrow, kt * 32, kA);
        v16h b = *(const v16h*)&sWg[(kt * 32 + lane) * 16];
        acc = __builtin_amdgcn_wmma_f32_16x16x32_f16(
            false, a, false, b, (short)0, acc, false, false);
    }
    const int mofs = (lane >> 4) * 8;
#pragma unroll
    for (int v = 0; v < 8; ++v) {
        const int row = rowBase + mofs + v;
        if (row < N) g[(size_t)row * CDIM + n] = acc[v];
    }
}

// ---------------------------------------------------------------------------
// Stage 4: per-node attention logits a_s = g.att_src, a_d = g.att_dst.
// ---------------------------------------------------------------------------
__global__ void k_att(const float* __restrict__ g, const float* __restrict__ att_s,
                      const float* __restrict__ att_d, float* __restrict__ a_s,
                      float* __restrict__ a_d, int N) {
    int i = blockIdx.x * blockDim.x + threadIdx.x;
    if (i >= N) return;
    const float* gr = g + (size_t)i * CDIM;
    float s = 0.f, d = 0.f;
#pragma unroll
    for (int c = 0; c < CDIM; ++c) { float v = gr[c]; s += v * att_s[c]; d += v * att_d[c]; }
    a_s[i] = s;
    a_d[i] = d;
}

// ---------------------------------------------------------------------------
// Stage 5: per-edge (incl. self loops) leaky-relu logit; segment max via
// ordered-uint atomicMax (global_atomic_max_u32).
// ---------------------------------------------------------------------------
__global__ void k_edge_max(const int* __restrict__ src, const int* __restrict__ dst,
                           const float* __restrict__ a_s, const float* __restrict__ a_d,
                           float* __restrict__ ebuf, unsigned* __restrict__ mmax,
                           int E, int N) {
    int t = blockIdx.x * blockDim.x + threadIdx.x;
    if (t >= E + N) return;
    int s = (t < E) ? src[t] : (t - E);
    int d = (t < E) ? dst[t] : (t - E);
    float ev = a_s[s] + a_d[d];
    ev = (ev > 0.0f) ? ev : NEG_SLOPE * ev;
    ebuf[t] = ev;
    atomicMax(&mmax[d], order_f32(ev));
}

// ---------------------------------------------------------------------------
// Stage 6: ex = exp(e - max[d]); denom[d] += ex. ebuf overwritten with ex.
// ---------------------------------------------------------------------------
__global__ void k_edge_exp(const int* __restrict__ dst, float* __restrict__ ebuf,
                           const unsigned* __restrict__ mmax, float* __restrict__ denom,
                           int E, int N) {
    int t = blockIdx.x * blockDim.x + threadIdx.x;
    if (t >= E + N) return;
    int d = (t < E) ? dst[t] : (t - E);
    float ex = expf(ebuf[t] - unorder_f32(mmax[d]));
    ebuf[t] = ex;
    unsafeAtomicAdd(&denom[d], ex);
}

// ---------------------------------------------------------------------------
// Stage 7: out[d] += (ex/denom[d]) * g[s].  One thread per (edge, channel).
// ---------------------------------------------------------------------------
__global__ void k_edge_agg(const int* __restrict__ src, const int* __restrict__ dst,
                           const float* __restrict__ ebuf, const float* __restrict__ denom,
                           const float* __restrict__ g, float* __restrict__ out,
                           int E, int N) {
    long long gid = (long long)blockIdx.x * blockDim.x + threadIdx.x;
    long long total = (long long)(E + N) * CDIM;
    if (gid >= total) return;
    int t = (int)(gid >> 4);
    int c = (int)(gid & 15);
    int s = (t < E) ? src[t] : (t - E);
    int d = (t < E) ? dst[t] : (t - E);
    float alpha = ebuf[t] / denom[d];
    unsafeAtomicAdd(&out[(size_t)d * CDIM + c], alpha * g[(size_t)s * CDIM + c]);
}

// ---------------------------------------------------------------------------
// Stage 8: out = log_softmax(out + b_gat) in place.
// ---------------------------------------------------------------------------
__global__ void k_logsoftmax(float* __restrict__ out, const float* __restrict__ bg, int N) {
    int i = blockIdx.x * blockDim.x + threadIdx.x;
    if (i >= N) return;
    float v[CDIM];
    float m = -3.402823466e38f;
#pragma unroll
    for (int c = 0; c < CDIM; ++c) {
        v[c] = out[(size_t)i * CDIM + c] + bg[c];
        m = fmaxf(m, v[c]);
    }
    float sum = 0.f;
#pragma unroll
    for (int c = 0; c < CDIM; ++c) sum += expf(v[c] - m);
    float ls = m + logf(sum);
#pragma unroll
    for (int c = 0; c < CDIM; ++c) out[(size_t)i * CDIM + c] = v[c] - ls;
}

// ---------------------------------------------------------------------------
// Host launcher.
// ---------------------------------------------------------------------------
extern "C" void kernel_launch(void* const* d_in, const int* in_sizes, int n_in,
                              void* d_out, int out_size, void* d_ws, size_t ws_size,
                              hipStream_t stream) {
    const float* x     = (const float*)d_in[0];
    const int*   ei    = (const int*)  d_in[1];
    const float* wl    = (const float*)d_in[2];
    const float* bl    = (const float*)d_in[3];
    const float* wr    = (const float*)d_in[4];
    const float* wg    = (const float*)d_in[5];
    const float* att_s = (const float*)d_in[6];
    const float* att_d = (const float*)d_in[7];
    const float* bg    = (const float*)d_in[8];
    float* out = (float*)d_out;

    const int N = in_sizes[0] / F_IN;
    const int E = in_sizes[1] / 2;
    const int* src = ei;
    const int* dst = ei + E;

    // Carve workspace (256B aligned slices).
    char* ws = (char*)d_ws;
    auto carve = [&](size_t bytes) {
        char* p = ws;
        ws += (bytes + 255) & ~(size_t)255;
        return (void*)p;
    };
    float*    agg   = (float*)   carve((size_t)N * F_IN * sizeof(float));
    _Float16* xh    = (_Float16*)carve((size_t)N * F_IN * sizeof(_Float16));
    _Float16* meanh = (_Float16*)carve((size_t)N * F_IN * sizeof(_Float16));
    _Float16* hbuf  = (_Float16*)carve((size_t)N * HDIM * sizeof(_Float16));
    float*    g     = (float*)   carve((size_t)N * CDIM * sizeof(float));
    float*    deg   = (float*)   carve((size_t)N * sizeof(float));
    float*    a_s   = (float*)   carve((size_t)N * sizeof(float));
    float*    a_d   = (float*)   carve((size_t)N * sizeof(float));
    unsigned* mmax  = (unsigned*)carve((size_t)N * sizeof(unsigned));
    float*    denom = (float*)   carve((size_t)N * sizeof(float));
    float*    ebuf  = (float*)   carve((size_t)(E + N) * sizeof(float));
    _Float16* wlf   = (_Float16*)carve(4096 * sizeof(_Float16));
    _Float16* wrf   = (_Float16*)carve(4096 * sizeof(_Float16));
    _Float16* wgf   = (_Float16*)carve(1024 * sizeof(_Float16));

    hipMemsetAsync(agg,   0, (size_t)N * F_IN * sizeof(float), stream);
    hipMemsetAsync(deg,   0, (size_t)N * sizeof(float), stream);
    hipMemsetAsync(mmax,  0, (size_t)N * sizeof(unsigned), stream);   // 0 < mapped(-inf)
    hipMemsetAsync(denom, 0, (size_t)N * sizeof(float), stream);
    hipMemsetAsync(out,   0, (size_t)N * CDIM * sizeof(float), stream);

    k_prep_w<<<(9216 + 255) / 256, 256, 0, stream>>>(wl, wr, wg, wlf, wrf, wgf);

    {   // scatter-mean
        long long total = (long long)E * F_IN;
        int blocks = (int)((total + 255) / 256);
        k_scatter<<<blocks, 256, 0, stream>>>(x, src, dst, agg, deg, E);
    }
    {
        long long total = (long long)N * F_IN;
        int blocks = (int)((total + 255) / 256);
        k_prep_feat<<<blocks, 256, 0, stream>>>(x, agg, deg, xh, meanh, N);
    }

    const int rblocks = (N + 63) / 64;   // 64 rows per 4-wave block
    k_sage_wmma<<<rblocks, 128, 0, stream>>>(xh, meanh, wlf, wrf, bl, hbuf, N);
    k_gat_wmma <<<rblocks, 128, 0, stream>>>(hbuf, wgf, g, N);
    k_att<<<(N + 255) / 256, 256, 0, stream>>>(g, att_s, att_d, a_s, a_d, N);

    const int ET = E + N;
    k_edge_max<<<(ET + 255) / 256, 256, 0, stream>>>(src, dst, a_s, a_d, ebuf, mmax, E, N);
    k_edge_exp<<<(ET + 255) / 256, 256, 0, stream>>>(dst, ebuf, mmax, denom, E, N);
    {
        long long total = (long long)ET * CDIM;
        int blocks = (int)((total + 255) / 256);
        k_edge_agg<<<blocks, 256, 0, stream>>>(src, dst, ebuf, denom, g, out, E, N);
    }
    k_logsoftmax<<<(N + 255) / 256, 256, 0, stream>>>(out, bg, N);
}
